// Model_3882650436638
// MI455X (gfx1250) — matrix operand order repaired and checked by reading, not compiled, not run on hardware
//
#include <hip/hip_runtime.h>

#define NUM_USERS 100000
#define NUM_JOBS  50000
#define NUM_EDGES 2000000
#define NUM_LABELS 500000
#define HID 128
#define USER_DIM 512
#define JOB_DIM 768
#define BN_EPS 1e-5f

typedef __attribute__((ext_vector_type(16))) _Float16 v16h;
typedef __attribute__((ext_vector_type(8)))  float    v8f;
typedef __attribute__((ext_vector_type(4)))  unsigned v4u;
typedef __attribute__((ext_vector_type(8)))  int      v8i;
typedef __attribute__((ext_vector_type(4)))  int      v4i;

// ---------------------------------------------------------------------------
// CDNA5 async global->LDS copy (ASYNCcnt-tracked), ISA 08_async_tensor §4.
// ---------------------------------------------------------------------------
__device__ __forceinline__ void async_load_b128(unsigned lds_off, const void* gptr) {
    asm volatile("global_load_async_to_lds_b128 %0, %1, off"
                 :: "v"(lds_off), "v"((unsigned long long)(uintptr_t)gptr)
                 : "memory");
}
__device__ __forceinline__ void wait_async0() {
    asm volatile("s_wait_asynccnt 0" ::: "memory");
}

// ---------------------------------------------------------------------------
// CDNA5 Tensor Data Mover: 2D tile (rows x row_elems f32, stride in elems)
// from global to LDS. Descriptor per ISA 08_async_tensor §8.3/§8.4:
//   g0: count=1 | lds_addr | global_addr[56:0] | type=2
//   g1: data_size=2 (4B) | tensor_dim0/1 | tile_dim0/1 | dim0_stride
//   remaining groups zero (2D tensor). Tracked by TENSORcnt.
// This toolchain exposes the 6-arg builtin (clang-23 / therock form).
// ---------------------------------------------------------------------------
__device__ __forceinline__ void tdm_load_2d(unsigned lds_addr, const void* gptr,
                                            int tile_rows, int row_elems,
                                            int row_stride_elems) {
    const unsigned long long ga = (unsigned long long)(uintptr_t)gptr;
    v4u g0;
    g0[0] = 1u;                                            // count=1, user desc
    g0[1] = lds_addr;                                      // LDS byte address
    g0[2] = (unsigned)(ga & 0xFFFFFFFFull);                // global_addr[31:0]
    g0[3] = (unsigned)((ga >> 32) & 0x01FFFFFFull)         // global_addr[56:32]
          | (2u << 30);                                    // type = 2 ("image")
    v8i g1;
    g1[0] = (int)(2u << 16);                               // data_size = 4B
    g1[1] = (int)((unsigned)(row_elems & 0xFFFF) << 16);   // tensor_dim0[15:0]
    g1[2] = (int)((unsigned)(tile_rows & 0xFFFF) << 16);   // tensor_dim1[15:0]
    g1[3] = (int)((unsigned)(row_elems & 0xFFFF) << 16);   // tile_dim0
    g1[4] = tile_rows & 0xFFFF;                            // tile_dim1
    g1[5] = row_stride_elems;                              // dim0_stride[31:0]
    g1[6] = 0;
    g1[7] = 0;
    const v4i z4 = {0, 0, 0, 0};
    const v8i z8 = {0, 0, 0, 0, 0, 0, 0, 0};
    __builtin_amdgcn_tensor_load_to_lds(g0, g1, z4, z4, z8, 0);
}

__device__ __forceinline__ void pack8(v16h& d, int base, float4 lo, float4 hi) {
    d[base + 0] = (_Float16)lo.x; d[base + 1] = (_Float16)lo.y;
    d[base + 2] = (_Float16)lo.z; d[base + 3] = (_Float16)lo.w;
    d[base + 4] = (_Float16)hi.x; d[base + 5] = (_Float16)hi.y;
    d[base + 6] = (_Float16)hi.z; d[base + 7] = (_Float16)hi.w;
}

// A fragment (16x32 f16, ISA 7.12.2): per lane two runs of 8 consecutive K
// values -> two pairs of float4 reads from the LDS-staged tile.
__device__ __forceinline__ v16h a_frag_lds(const float* tile, int ldk, int k0, int lane) {
    const int m = lane & 15, half = lane >> 4;
    const float* row = tile + m * ldk;
    const float4* p0 = (const float4*)(row + k0 + half * 8);
    const float4* p1 = (const float4*)(row + k0 + 16 + half * 8);
    const float4 x0 = p0[0], x1 = p0[1], y0 = p1[0], y1 = p1[1];
    v16h a;
    pack8(a, 0, x0, x1);
    pack8(a, 8, y0, y1);
    return a;
}

// B fragment (32x16 f16 = W^T, W row-major HID x K): per lane two runs of 8
// consecutive K values -> 4 x global_load_b128 (weights stay L2-resident).
__device__ __forceinline__ v16h b_frag_g(const float* __restrict__ W, int ldw,
                                         int nBase, int k0, int lane) {
    const int n = lane & 15, half = lane >> 4;
    const float4* p = (const float4*)(W + (size_t)(nBase + n) * (size_t)ldw + k0 + half * 8);
    const float4 x0 = p[0], x1 = p[1], x2 = p[4], x3 = p[5];
    v16h b;
    pack8(b, 0, x0, x1);
    pack8(b, 8, x2, x3);
    return b;
}

// ---------------------------------------------------------------------------
// Encoder: Y = relu(BN(X @ W^T + bias)); one wave per 16x16 tile, block (32,8)
// covers the full N=128 strip; the 16-row A tile is staged once per block into
// LDS with async b128 DMA and shared by all 8 waves.
// ---------------------------------------------------------------------------
template <int K>
__global__ void __launch_bounds__(256)
encoder_kernel(const float* __restrict__ X, const float* __restrict__ W,
               const float* __restrict__ bias,
               const float* __restrict__ gamma, const float* __restrict__ beta,
               const float* __restrict__ mean,  const float* __restrict__ var,
               float* __restrict__ Y) {
    __shared__ float atile[16 * K];
    const int lane  = threadIdx.x;
    const int nBase = threadIdx.y * 16;
    const int mBase = blockIdx.x * 16;
    const int tid   = threadIdx.y * 32 + threadIdx.x;

    const unsigned ldsA = (unsigned)(uintptr_t)&atile[0];
    constexpr int NV4 = 16 * K / 4;              // float4 elements in tile
#pragma unroll
    for (int i = tid; i < NV4; i += 256) {       // NV4 % 256 == 0: no divergence
        const int r = i / (K / 4);
        const int c = i - r * (K / 4);
        async_load_b128(ldsA + (unsigned)i * 16,
                        (const float4*)(X + (size_t)(mBase + r) * K) + c);
    }
    wait_async0();
    __syncthreads();

    v8f acc = {};
#pragma unroll 4
    for (int k0 = 0; k0 < K; k0 += 32) {
        const v16h a = a_frag_lds(atile, K, k0, lane);
        const v16h b = b_frag_g(W, K, nBase, k0, lane);
        acc = __builtin_amdgcn_wmma_f32_16x16x32_f16(false, a, false, b,
                                                     (short)0, acc, false, false);
    }

    const int n     = nBase + (lane & 15);
    const int half  = lane >> 4;
    const float scl = gamma[n] * rsqrtf(var[n] + BN_EPS);
    const float sh  = (bias[n] - mean[n]) * scl + beta[n];
#pragma unroll
    for (int r = 0; r < 8; ++r) {
        const int m = mBase + r + 8 * half;
        Y[(size_t)m * HID + n] = fmaxf(acc[r] * scl + sh, 0.0f);
    }
}

// ---------------------------------------------------------------------------
// SAGE: Y = act(AGG @ Wl^T + bl + XD @ Wr^T), K = 128. Both 16x128 A tiles
// fetched by the Tensor Data Mover (one descriptor each, wave 0 issues),
// then shared by all 8 waves.
// ---------------------------------------------------------------------------
__global__ void __launch_bounds__(256)
sage_kernel(const float* __restrict__ AGG, const float* __restrict__ XD,
            const float* __restrict__ Wl,  const float* __restrict__ bl,
            const float* __restrict__ Wr,  float* __restrict__ Y, int do_relu) {
    __shared__ float atile[16 * HID];
    __shared__ float xtile[16 * HID];
    const int lane  = threadIdx.x;
    const int nBase = threadIdx.y * 16;
    const int mBase = blockIdx.x * 16;

    if (threadIdx.y == 0) {   // one wave programs the TDM (EXEC-independent op)
        tdm_load_2d((unsigned)(uintptr_t)&atile[0], AGG + (size_t)mBase * HID,
                    16, HID, HID);
        tdm_load_2d((unsigned)(uintptr_t)&xtile[0], XD + (size_t)mBase * HID,
                    16, HID, HID);
        __builtin_amdgcn_s_wait_tensorcnt(0);
    }
    __syncthreads();

    v8f acc = {};
#pragma unroll
    for (int k0 = 0; k0 < HID; k0 += 32) {
        v16h a = a_frag_lds(atile, HID, k0, lane);
        v16h b = b_frag_g(Wl, HID, nBase, k0, lane);
        acc = __builtin_amdgcn_wmma_f32_16x16x32_f16(false, a, false, b,
                                                     (short)0, acc, false, false);
        a = a_frag_lds(xtile, HID, k0, lane);
        b = b_frag_g(Wr, HID, nBase, k0, lane);
        acc = __builtin_amdgcn_wmma_f32_16x16x32_f16(false, a, false, b,
                                                     (short)0, acc, false, false);
    }

    const int n    = nBase + (lane & 15);
    const int half = lane >> 4;
    const float bb = bl[n];
#pragma unroll
    for (int r = 0; r < 8; ++r) {
        const int m = mBase + r + 8 * half;
        float v = acc[r] + bb;
        if (do_relu) v = fmaxf(v, 0.0f);
        Y[(size_t)m * HID + n] = v;
    }
}

// ---------------------------------------------------------------------------
// Edge-degree counts (scatter +1.0 onto dst)
// ---------------------------------------------------------------------------
__global__ void count_kernel(const int* __restrict__ idx, float* __restrict__ cnt, int n) {
    const int t = blockIdx.x * blockDim.x + threadIdx.x;
    if (t < n) unsafeAtomicAdd(&cnt[idx[t]], 1.0f);
}

// ---------------------------------------------------------------------------
// Scatter-add: one wave per edge; each lane moves 4 contiguous features
// ---------------------------------------------------------------------------
__global__ void scatter_kernel(const float* __restrict__ X, const int* __restrict__ src,
                               const int* __restrict__ dst, float* __restrict__ AGG,
                               int nEdges) {
    const long long t = (long long)blockIdx.x * blockDim.x + threadIdx.x;
    const int e    = (int)(t >> 5);
    const int lane = (int)(t & 31);
    if (e >= nEdges) return;
    const float4 v = ((const float4*)(X + (size_t)src[e] * HID))[lane];
    float* ag = AGG + (size_t)dst[e] * HID + 4 * lane;
    unsafeAtomicAdd(ag + 0, v.x);
    unsafeAtomicAdd(ag + 1, v.y);
    unsafeAtomicAdd(ag + 2, v.z);
    unsafeAtomicAdd(ag + 3, v.w);
}

// ---------------------------------------------------------------------------
// Mean normalization: AGG[node] *= 1/max(cnt[node],1)
// ---------------------------------------------------------------------------
__global__ void normalize_kernel(float* __restrict__ AGG, const float* __restrict__ cnt,
                                 int nNodes) {
    const long long t = (long long)blockIdx.x * blockDim.x + threadIdx.x;
    const int node = (int)(t >> 5);
    const int lane = (int)(t & 31);
    if (node >= nNodes) return;
    const float r = 1.0f / fmaxf(cnt[node], 1.0f);
    float4* p = (float4*)(AGG + (size_t)node * HID);
    float4 v = p[lane];
    v.x *= r; v.y *= r; v.z *= r; v.w *= r;
    p[lane] = v;
}

// ---------------------------------------------------------------------------
// Link classifier: out[e] = dot(U2[ls[e]], J2[ld[e]]), wave per edge
// ---------------------------------------------------------------------------
__global__ void dot_kernel(const float* __restrict__ U, const float* __restrict__ J,
                           const int* __restrict__ ls, const int* __restrict__ ld,
                           float* __restrict__ out, int nL) {
    const long long t = (long long)blockIdx.x * blockDim.x + threadIdx.x;
    const int e    = (int)(t >> 5);
    const int lane = (int)(t & 31);
    if (e >= nL) return;
    const float4 a = ((const float4*)(U + (size_t)ls[e] * HID))[lane];
    const float4 b = ((const float4*)(J + (size_t)ld[e] * HID))[lane];
    float s = a.x * b.x + a.y * b.y + a.z * b.z + a.w * b.w;
#pragma unroll
    for (int off = 16; off > 0; off >>= 1) s += __shfl_down(s, off, 32);
    if (lane == 0) out[e] = s;
}

// ---------------------------------------------------------------------------
extern "C" void kernel_launch(void* const* d_in, const int* in_sizes, int n_in,
                              void* d_out, int out_size, void* d_ws, size_t ws_size,
                              hipStream_t stream) {
    const float* user_x     = (const float*)d_in[0];
    const float* job_x      = (const float*)d_in[1];
    const float* user_W     = (const float*)d_in[2];
    const float* user_b     = (const float*)d_in[3];
    const float* user_gamma = (const float*)d_in[4];
    const float* user_beta  = (const float*)d_in[5];
    const float* user_mean  = (const float*)d_in[6];
    const float* user_var   = (const float*)d_in[7];
    const float* job_W      = (const float*)d_in[8];
    const float* job_b      = (const float*)d_in[9];
    const float* job_gamma  = (const float*)d_in[10];
    const float* job_beta   = (const float*)d_in[11];
    const float* job_mean   = (const float*)d_in[12];
    const float* job_var    = (const float*)d_in[13];
    const float* l1_rates_Wl = (const float*)d_in[14];
    const float* l1_rates_Wr = (const float*)d_in[15];
    const float* l1_rev_Wl   = (const float*)d_in[16];
    const float* l1_rev_Wr   = (const float*)d_in[17];
    const float* l2_rates_Wl = (const float*)d_in[18];
    const float* l2_rates_Wr = (const float*)d_in[19];
    const float* l2_rev_Wl   = (const float*)d_in[20];
    const float* l2_rev_Wr   = (const float*)d_in[21];
    const float* l1_rates_bl = (const float*)d_in[22];
    const float* l1_rev_bl   = (const float*)d_in[23];
    const float* l2_rates_bl = (const float*)d_in[24];
    const float* l2_rev_bl   = (const float*)d_in[25];
    const int* rates_src = (const int*)d_in[26];
    const int* rates_dst = (const int*)d_in[27];
    const int* label_src = (const int*)d_in[28];
    const int* label_dst = (const int*)d_in[29];

    float* ws = (float*)d_ws;
    size_t o = 0;
    float* u    = ws + o; o += (size_t)NUM_USERS * HID;   // reused for u2
    float* j    = ws + o; o += (size_t)NUM_JOBS  * HID;   // reused for j2
    float* u1   = ws + o; o += (size_t)NUM_USERS * HID;
    float* j1   = ws + o; o += (size_t)NUM_JOBS  * HID;
    float* aggU = ws + o; o += (size_t)NUM_USERS * HID;
    float* aggJ = ws + o; o += (size_t)NUM_JOBS  * HID;
    float* cntU = ws + o; o += (size_t)NUM_USERS;
    float* cntJ = ws + o; o += (size_t)NUM_JOBS;

    const dim3 gemmBlk(32, 8);
    const int userTiles = NUM_USERS / 16;  // 6250
    const int jobTiles  = NUM_JOBS / 16;   // 3125
    const int edgeBlks  = (int)(((long long)NUM_EDGES * 32 + 255) / 256);
    const int userNBlks = (int)(((long long)NUM_USERS * 32 + 255) / 256);
    const int jobNBlks  = (int)(((long long)NUM_JOBS  * 32 + 255) / 256);
    const int lblBlks   = (int)(((long long)NUM_LABELS * 32 + 255) / 256);

    // degree counts (shared by both layers)
    (void)hipMemsetAsync(cntU, 0, sizeof(float) * NUM_USERS, stream);
    (void)hipMemsetAsync(cntJ, 0, sizeof(float) * NUM_JOBS, stream);
    count_kernel<<<(NUM_EDGES + 255) / 256, 256, 0, stream>>>(rates_src, cntU, NUM_EDGES);
    count_kernel<<<(NUM_EDGES + 255) / 256, 256, 0, stream>>>(rates_dst, cntJ, NUM_EDGES);

    // node encoders
    encoder_kernel<USER_DIM><<<userTiles, gemmBlk, 0, stream>>>(
        user_x, user_W, user_b, user_gamma, user_beta, user_mean, user_var, u);
    encoder_kernel<JOB_DIM><<<jobTiles, gemmBlk, 0, stream>>>(
        job_x, job_W, job_b, job_gamma, job_beta, job_mean, job_var, j);

    // layer 1: rates (u -> jobs)
    (void)hipMemsetAsync(aggJ, 0, sizeof(float) * (size_t)NUM_JOBS * HID, stream);
    scatter_kernel<<<edgeBlks, 256, 0, stream>>>(u, rates_src, rates_dst, aggJ, NUM_EDGES);
    normalize_kernel<<<jobNBlks, 256, 0, stream>>>(aggJ, cntJ, NUM_JOBS);
    sage_kernel<<<jobTiles, gemmBlk, 0, stream>>>(aggJ, j, l1_rates_Wl, l1_rates_bl,
                                                  l1_rates_Wr, j1, 1);

    // layer 1: rev_rates (j -> users)
    (void)hipMemsetAsync(aggU, 0, sizeof(float) * (size_t)NUM_USERS * HID, stream);
    scatter_kernel<<<edgeBlks, 256, 0, stream>>>(j, rates_dst, rates_src, aggU, NUM_EDGES);
    normalize_kernel<<<userNBlks, 256, 0, stream>>>(aggU, cntU, NUM_USERS);
    sage_kernel<<<userTiles, gemmBlk, 0, stream>>>(aggU, u, l1_rev_Wl, l1_rev_bl,
                                                   l1_rev_Wr, u1, 1);

    // layer 2: rates (u1 -> jobs) -> j2 (stored in j buffer, no relu)
    (void)hipMemsetAsync(aggJ, 0, sizeof(float) * (size_t)NUM_JOBS * HID, stream);
    scatter_kernel<<<edgeBlks, 256, 0, stream>>>(u1, rates_src, rates_dst, aggJ, NUM_EDGES);
    normalize_kernel<<<jobNBlks, 256, 0, stream>>>(aggJ, cntJ, NUM_JOBS);
    sage_kernel<<<jobTiles, gemmBlk, 0, stream>>>(aggJ, j1, l2_rates_Wl, l2_rates_bl,
                                                  l2_rates_Wr, j, 0);

    // layer 2: rev_rates (j1 -> users) -> u2 (stored in u buffer, no relu)
    (void)hipMemsetAsync(aggU, 0, sizeof(float) * (size_t)NUM_USERS * HID, stream);
    scatter_kernel<<<edgeBlks, 256, 0, stream>>>(j1, rates_dst, rates_src, aggU, NUM_EDGES);
    normalize_kernel<<<userNBlks, 256, 0, stream>>>(aggU, cntU, NUM_USERS);
    sage_kernel<<<userTiles, gemmBlk, 0, stream>>>(aggU, u1, l2_rev_Wl, l2_rev_bl,
                                                   l2_rev_Wr, u, 0);

    // link classifier
    dot_kernel<<<lblBlks, 256, 0, stream>>>(u, j, label_src, label_dst,
                                            (float*)d_out, NUM_LABELS);
}
